// GWNet2_65420941853388
// MI455X (gfx1250) — matrix operand order repaired and checked by reading, not compiled, not run on hardware
//
#include <hip/hip_runtime.h>
#include <math.h>

// ---------------------------------------------------------------------------
// Graph-WaveNet forward for MI455X (gfx1250, wave32).
// Intermediates: (B, C, L, N) row-major, node dim innermost.
// Dominant graph-conv einsum  x1 = h @ adp  is a row-major GEMM
//   C[M x 1024] = A[M x 1024] * adp[1024 x 1024],  M = 2048*l (mult of 128)
// executed with V_WMMA_F32_16X16X32_F16 (f16 inputs, f32 accumulation).
// Residual chain / skip path / epilogues stay fp32.
// ---------------------------------------------------------------------------

typedef __attribute__((ext_vector_type(16))) _Float16 v16h;
typedef __attribute__((ext_vector_type(8)))  _Float16 h8;
typedef __attribute__((ext_vector_type(8)))  float    v8f;

#define Bx 64
#define Nn 1024
#define RC 32
#define SC 256
#define EC 512
#define OUTC 12
#define APT 10

// ------- adjacency: softmax(relu(nv1 @ nv2), axis=1), stored TRANSPOSED f16 -
// adpT[w][v] = adp[v][w]  so GEMM B-fragments read contiguous K-runs.
__global__ __launch_bounds__(256) void adp_kernel(
    const float* __restrict__ nv1, const float* __restrict__ nv2,
    _Float16* __restrict__ adpT) {
  const int v = blockIdx.x;
  const int t = threadIdx.x;
  __shared__ float red[256];

  float nrow[APT];
#pragma unroll
  for (int k = 0; k < APT; ++k) nrow[k] = nv1[v * APT + k];

  float vals[4];
#pragma unroll
  for (int j = 0; j < 4; ++j) {
    const int w = t + j * 256;
    float s = 0.f;
#pragma unroll
    for (int k = 0; k < APT; ++k) s += nrow[k] * nv2[k * Nn + w];
    vals[j] = fmaxf(s, 0.f);
  }
  float lm = fmaxf(fmaxf(vals[0], vals[1]), fmaxf(vals[2], vals[3]));
  red[t] = lm;
  __syncthreads();
  for (int s = 128; s > 0; s >>= 1) {
    if (t < s) red[t] = fmaxf(red[t], red[t + s]);
    __syncthreads();
  }
  const float rowmax = red[0];
  __syncthreads();
  float ls = 0.f;
#pragma unroll
  for (int j = 0; j < 4; ++j) {
    vals[j] = expf(vals[j] - rowmax);
    ls += vals[j];
  }
  red[t] = ls;
  __syncthreads();
  for (int s = 128; s > 0; s >>= 1) {
    if (t < s) red[t] += red[t + s];
    __syncthreads();
  }
  const float inv = 1.f / red[0];
#pragma unroll
  for (int j = 0; j < 4; ++j) {
    const int w = t + j * 256;
    adpT[(size_t)w * Nn + v] = (_Float16)(vals[j] * inv);
  }
}

// ---------------- start conv: x (B,2,N,13) -> h (B,32,13,N) -----------------
__global__ __launch_bounds__(256) void start_conv_kernel(
    const float* __restrict__ x, const float* __restrict__ w,
    const float* __restrict__ bias, float* __restrict__ h) {
  const int v = blockIdx.x * 256 + threadIdx.x;
  const int o = blockIdx.y / 13;
  const int l = blockIdx.y % 13;
  const int b = blockIdx.z;
  const float x0 = x[((size_t)(b * 2 + 0) * Nn + v) * 13 + l];
  const float x1 = x[((size_t)(b * 2 + 1) * Nn + v) * 13 + l];
  h[((size_t)(b * RC + o) * 13 + l) * Nn + v] =
      w[o * 2 + 0] * x0 + w[o * 2 + 1] * x1 + bias[o];
}

// ------- fused dilated filter/gate conv + tanh*sigmoid (f32 + f16 copy) -----
__global__ __launch_bounds__(256) void dconv_kernel(
    const float* __restrict__ h, float* __restrict__ gated,
    _Float16* __restrict__ gated16, const float* __restrict__ fw,
    const float* __restrict__ fb, const float* __restrict__ gw,
    const float* __restrict__ gb, int l_in, int l_out, int d) {
  const int v = blockIdx.x * 256 + threadIdx.x;
  const int o = blockIdx.y / l_out;
  const int ll = blockIdx.y % l_out;
  const int b = blockIdx.z;

  const size_t base = ((size_t)(b * RC) * l_in + ll) * Nn + v;
  const size_t cs = (size_t)l_in * Nn;
  float f = fb[o];
  float g = gb[o];
#pragma unroll
  for (int c = 0; c < RC; ++c) {
    const float a0 = h[base + c * cs];
    const float a1 = h[base + c * cs + (size_t)d * Nn];
    const int wi = (o * RC + c) * 2;
    f += fw[wi] * a0 + fw[wi + 1] * a1;
    g += gw[wi] * a0 + gw[wi + 1] * a1;
  }
  const float out = tanhf(f) * (1.f / (1.f + expf(-g)));
  const size_t oi = ((size_t)(b * RC + o) * l_out + ll) * Nn + v;
  gated[oi] = out;
  gated16[oi] = (_Float16)out;
}

// ------ skip conv at the LAST time position only, accumulated over layers ---
__global__ __launch_bounds__(256) void skip_kernel(
    const float* __restrict__ gated, const float* __restrict__ w,
    const float* __restrict__ bias, float* __restrict__ skip_acc,
    int l_out, int init) {
  const int v = blockIdx.x * 256 + threadIdx.x;
  const int o = blockIdx.y;  // 0..255
  const int b = blockIdx.z;
  const size_t base = ((size_t)(b * RC) * l_out + (l_out - 1)) * Nn + v;
  const size_t cs = (size_t)l_out * Nn;
  float s = bias[o];
#pragma unroll
  for (int c = 0; c < RC; ++c) s += w[o * RC + c] * gated[base + c * cs];
  const size_t oi = ((size_t)b * SC + o) * Nn + v;
  skip_acc[oi] = init ? s : (skip_acc[oi] + s);
}

// ------------- WMMA f16 GEMM:  C16[M x 1024] = A16[M x 1024] * adpT' --------
// BT is the TRANSPOSED B (BT[n][k]), so B-fragments are contiguous K-runs.
// Block: 256 threads = 8 waves, tile 128(M) x 128(N), K-step 32.
// Waves 2(M) x 4(N); each wave owns 4x2 16x16 tiles (f32 accumulators).
__global__ __launch_bounds__(256) void graph_gemm_f16_kernel(
    const _Float16* __restrict__ A, const _Float16* __restrict__ BT,
    _Float16* __restrict__ C, int M) {
  (void)M;  // grid sized exactly; M is a multiple of 128
  __shared__ _Float16 As[128][40];  // 80B pitch: 16B-aligned chunks
  __shared__ _Float16 Bs[128][40];

  const int tid = threadIdx.x;
  const int lane = tid & 31;
  const int wv = tid >> 5;
  const int waveM = wv & 1;   // 0..1
  const int waveN = wv >> 1;  // 0..3
  const int m16 = lane & 15;
  const int half = lane >> 4;
  const int blockN = blockIdx.x * 128;
  const int blockM = blockIdx.y * 128;

  v8f acc[4][2] = {};

  const int sRow = tid >> 1;        // 0..127
  const int sC = (tid & 1) * 16;    // 0 or 16 halves

  const _Float16* Arow = A + (size_t)(blockM + sRow) * Nn + sC;
  const _Float16* Brow = BT + (size_t)(blockN + sRow) * Nn + sC;

  for (int k0 = 0; k0 < Nn; k0 += 32) {
    *(h8*)&As[sRow][sC] = *(const h8*)(Arow + k0);
    *(h8*)&As[sRow][sC + 8] = *(const h8*)(Arow + k0 + 8);
    *(h8*)&Bs[sRow][sC] = *(const h8*)(Brow + k0);
    *(h8*)&Bs[sRow][sC + 8] = *(const h8*)(Brow + k0 + 8);
    __syncthreads();

    v16h av[4], bv[2];
#pragma unroll
    for (int mi = 0; mi < 4; ++mi) {
      // A(m,k) f16 layout: elem j<8 -> k = 8*half + j ; j>=8 -> 16+8*half+j-8
      const int r = waveM * 64 + mi * 16 + m16;
      const h8 a0 = *(const h8*)&As[r][8 * half];
      const h8 a1 = *(const h8*)&As[r][16 + 8 * half];
      av[mi] = __builtin_shufflevector(a0, a1, 0, 1, 2, 3, 4, 5, 6, 7, 8, 9,
                                       10, 11, 12, 13, 14, 15);
    }
#pragma unroll
    for (int nj = 0; nj < 2; ++nj) {
      // B(k,n) f16 layout: elem j<8 -> k = 8*half + j ; j>=8 -> 16+8*half+j-8
      const int cb = waveN * 32 + nj * 16 + m16;
      const h8 b0 = *(const h8*)&Bs[cb][8 * half];
      const h8 b1 = *(const h8*)&Bs[cb][16 + 8 * half];
      bv[nj] = __builtin_shufflevector(b0, b1, 0, 1, 2, 3, 4, 5, 6, 7, 8, 9,
                                       10, 11, 12, 13, 14, 15);
    }
#pragma unroll
    for (int mi = 0; mi < 4; ++mi)
#pragma unroll
      for (int nj = 0; nj < 2; ++nj)
        acc[mi][nj] = __builtin_amdgcn_wmma_f32_16x16x32_f16(
            false, av[mi], false, bv[nj], (short)0, acc[mi][nj], false, false);
    __syncthreads();
  }

  // C/D layout: reg r holds M=r (lanes 0-15) and M=r+8 (lanes 16-31); N=lane%16
#pragma unroll
  for (int mi = 0; mi < 4; ++mi)
#pragma unroll
    for (int nj = 0; nj < 2; ++nj)
#pragma unroll
      for (int r = 0; r < 8; ++r) {
        const int m = blockM + waveM * 64 + mi * 16 + r + 8 * half;
        const int n = blockN + waveN * 32 + nj * 16 + m16;
        C[(size_t)m * Nn + n] = (_Float16)acc[mi][nj][r];
      }
}

// -------- gc conv over concat(gated,x1,x2) + residual + batchnorm -----------
__global__ __launch_bounds__(256) void gc_kernel(
    const float* __restrict__ gated, const _Float16* __restrict__ x1,
    const _Float16* __restrict__ x2, const float* __restrict__ residual,
    float* __restrict__ h_out, const float* __restrict__ w,
    const float* __restrict__ bias, const float* __restrict__ bn_g,
    const float* __restrict__ bn_b, int l_in, int l_out, int d) {
  const int v = blockIdx.x * 256 + threadIdx.x;
  const int o = blockIdx.y / l_out;
  const int ll = blockIdx.y % l_out;
  const int b = blockIdx.z;

  const size_t base = ((size_t)(b * RC) * l_out + ll) * Nn + v;
  const size_t cs = (size_t)l_out * Nn;
  float g = bias[o];
#pragma unroll
  for (int c = 0; c < RC; ++c) {
    g += w[o * 96 + c] * gated[base + c * cs];
    g += w[o * 96 + 32 + c] * (float)x1[base + c * cs];
    g += w[o * 96 + 64 + c] * (float)x2[base + c * cs];
  }
  const float res = residual[((size_t)(b * RC + o) * l_in + (ll + d)) * Nn + v];
  const float scale = bn_g[o] * rsqrtf(1.f + 1e-5f);
  h_out[((size_t)(b * RC + o) * l_out + ll) * Nn + v] =
      (g + res) * scale + bn_b[o];
}

// ---------------- end convs -------------------------------------------------
__global__ __launch_bounds__(256) void end1_kernel(
    const float* __restrict__ skip_acc, const float* __restrict__ w,
    const float* __restrict__ bias, float* __restrict__ e1) {
  const int v = blockIdx.x * 256 + threadIdx.x;
  const int o = blockIdx.y;  // 0..511
  const int b = blockIdx.z;
  float s = bias[o];
#pragma unroll 8
  for (int c = 0; c < SC; ++c)
    s += w[o * SC + c] * fmaxf(skip_acc[((size_t)b * SC + c) * Nn + v], 0.f);
  e1[((size_t)b * EC + o) * Nn + v] = fmaxf(s, 0.f);
}

__global__ __launch_bounds__(256) void end2_kernel(
    const float* __restrict__ e1, const float* __restrict__ w,
    const float* __restrict__ bias, float* __restrict__ out) {
  const int v = blockIdx.x * 256 + threadIdx.x;
  const int o = blockIdx.y;  // 0..11
  const int b = blockIdx.z;
  float s = bias[o];
#pragma unroll 8
  for (int c = 0; c < EC; ++c)
    s += w[o * EC + c] * e1[((size_t)b * EC + c) * Nn + v];
  out[((size_t)b * OUTC + o) * Nn + v] = s;
}

// ---------------------------------------------------------------------------
extern "C" void kernel_launch(void* const* d_in, const int* in_sizes, int n_in,
                              void* d_out, int out_size, void* d_ws,
                              size_t ws_size, hipStream_t stream) {
  (void)in_sizes; (void)n_in; (void)out_size; (void)ws_size;
  const float* x        = (const float*)d_in[0];
  const float* start_w  = (const float*)d_in[1];
  const float* start_b  = (const float*)d_in[2];
  const float* nv1      = (const float*)d_in[3];
  const float* nv2      = (const float*)d_in[4];
  const float* filter_w = (const float*)d_in[5];
  const float* filter_b = (const float*)d_in[6];
  const float* gate_w   = (const float*)d_in[7];
  const float* gate_b   = (const float*)d_in[8];
  const float* skip_w   = (const float*)d_in[9];
  const float* skip_b   = (const float*)d_in[10];
  const float* gc_w     = (const float*)d_in[11];
  const float* gc_b     = (const float*)d_in[12];
  const float* bn_g     = (const float*)d_in[13];
  const float* bn_b     = (const float*)d_in[14];
  const float* end1_w   = (const float*)d_in[15];
  const float* end1_b   = (const float*)d_in[16];
  const float* end2_w   = (const float*)d_in[17];
  const float* end2_b   = (const float*)d_in[18];

  // Workspace carve (bytes). Total ~560 MB.
  const size_t HB = (size_t)Bx * RC * 13 * Nn;  // 27,262,976 elements
  char* base = (char*)d_ws;
  size_t off = 0;
  auto alloc = [&](size_t bytes) -> void* {
    void* p = base + off;
    off += (bytes + 255) & ~(size_t)255;
    return p;
  };
  _Float16* adpT = (_Float16*)alloc(2ull * Nn * Nn);  // transposed f16 adp
  float* bA      = (float*)alloc(4ull * HB);          // h / residual (ping)
  float* bB      = (float*)alloc(4ull * HB);          // h next (pong)
  float* g32     = (float*)alloc(4ull * HB);          // gated f32
  _Float16* g16  = (_Float16*)alloc(2ull * HB);       // gated f16 (GEMM A)
  _Float16* x1h  = (_Float16*)alloc(2ull * HB);       // x1 f16
  _Float16* x2h  = (_Float16*)alloc(2ull * HB);       // x2 f16
  float* skipA   = (float*)alloc(4ull * (size_t)Bx * SC * Nn);
  float* e1      = g32;  // reuse g32+g16 region (163.6MB >= 134.2MB needed)

  const dim3 blk(256);

  adp_kernel<<<dim3(Nn), blk, 0, stream>>>(nv1, nv2, adpT);
  start_conv_kernel<<<dim3(4, RC * 13, Bx), blk, 0, stream>>>(x, start_w,
                                                              start_b, bA);

  static const int DILS[8] = {1, 2, 1, 2, 1, 2, 1, 2};
  int l = 13;
  float* cur = bA;
  float* nxt = bB;
  for (int i = 0; i < 8; ++i) {
    const int d = DILS[i];
    const int lo = l - d;
    dconv_kernel<<<dim3(4, RC * lo, Bx), blk, 0, stream>>>(
        cur, g32, g16, filter_w + (size_t)i * RC * RC * 2, filter_b + i * RC,
        gate_w + (size_t)i * RC * RC * 2, gate_b + i * RC, l, lo, d);
    skip_kernel<<<dim3(4, SC, Bx), blk, 0, stream>>>(
        g32, skip_w + (size_t)i * SC * RC, skip_b + i * SC, skipA, lo,
        i == 0 ? 1 : 0);
    if (i == 7) break;
    const int M = Bx * RC * lo;  // 2048*lo, multiple of 128
    graph_gemm_f16_kernel<<<dim3(8, M / 128), blk, 0, stream>>>(g16, adpT,
                                                                x1h, M);
    graph_gemm_f16_kernel<<<dim3(8, M / 128), blk, 0, stream>>>(x1h, adpT,
                                                                x2h, M);
    gc_kernel<<<dim3(4, RC * lo, Bx), blk, 0, stream>>>(
        g32, x1h, x2h, cur, nxt, gc_w + (size_t)i * RC * 96, gc_b + i * RC,
        bn_g + i * RC, bn_b + i * RC, l, lo, d);
    float* t = cur; cur = nxt; nxt = t;
    l = lo;
  }

  end1_kernel<<<dim3(4, EC, Bx), blk, 0, stream>>>(skipA, end1_w, end1_b, e1);
  end2_kernel<<<dim3(4, OUTC, Bx), blk, 0, stream>>>(e1, end2_w, end2_b,
                                                     (float*)d_out);
}
